// FRCNN_36137854828566
// MI455X (gfx1250) — compile-verified, matrix operand order checked by reading, not run on hardware
//
#include <hip/hip_runtime.h>

// ---------------------------------------------------------------------------
// Faster R-CNN forward for MI455X (gfx1250): f16 WMMA for conv tower + FC
// stack, TDM (tensor_load_to_lds) double-buffered LDS staging for weights,
// scalar kernels for the irregular glue.
// ---------------------------------------------------------------------------

typedef __attribute__((ext_vector_type(16))) _Float16 v16h;
typedef __attribute__((ext_vector_type(8)))  float    v8f;
typedef __attribute__((ext_vector_type(4)))  unsigned u32x4;
typedef __attribute__((ext_vector_type(8)))  int      i32x8;
typedef __attribute__((ext_vector_type(4)))  int      i32x4;
typedef unsigned long long ull;

#define IMG_H 600
#define IMG_W 800
#define FH 37
#define FW 50
#define NPIX (FH * FW)        // 1850
#define NA   (NPIX * 9)       // 16650
#define NMS_K 2000
#define RDIM  (NMS_K + 8)     // rois + gt
#define NSAMPLE 128

// output layout (flat f32, concatenated in reference return order)
#define O_RPN_CLS    0
#define O_RPN_REG    33300
#define O_PRED_CLS   99900
#define O_PRED_REG   102588
#define O_TG_RPN_CLS 103100
#define O_TG_RPN_REG 119750
#define O_TG_CLS     186350
#define O_TG_REG     186478

// compiler fence: orders LDS reads after TDM waits and stops the compiler
// from reasoning about LDS contents it never saw stored (TDM writes).
#define LDS_FENCE() asm volatile("" ::: "memory")

__device__ __forceinline__ int iclampi(int v, int lo, int hi) {
    return v < lo ? lo : (v > hi ? hi : v);
}

// paired f16 load from LDS (two contiguous halves -> fragment elems e, e+1);
// plain load so addrspace inference emits ds_load_b32.
__device__ __forceinline__ void ld2h(const _Float16* p, v16h& f, int e) {
    _Float16 h2[2];
    __builtin_memcpy(h2, p, 4);   // 4B-aligned by construction
    f[e] = h2[0];
    f[e + 1] = h2[1];
}

// ---------------------------------------------------------------------------
// TDM helper: 2D tile load (tile_h rows x tile_w f16), global row stride
// row_stride_elems, into LDS byte offset lds_byte_off (D# group0+group1).
// ---------------------------------------------------------------------------
#if __has_builtin(__builtin_amdgcn_tensor_load_to_lds)
#define HAVE_TDM 1
#else
#define HAVE_TDM 0
#endif

#if HAVE_TDM
#if __has_builtin(__builtin_amdgcn_s_wait_tensorcnt)
#define WAIT_TC(n) __builtin_amdgcn_s_wait_tensorcnt(n)
#else
#define WAIT_TC(n) asm volatile("s_wait_tensorcnt " #n ::: "memory")
#endif

__device__ __forceinline__ void tdm_load_2d_f16(unsigned lds_byte_off,
                                                const _Float16* g, unsigned tile_w,
                                                unsigned tile_h, ull row_stride_elems,
                                                ull tdim0, ull tdim1) {
    ull ga = (ull)(const void*)g;
    u32x4 g0 = {1u,                                     // count=1 valid descriptor
                lds_byte_off,                           // lds_addr
                (unsigned)ga,                           // global_addr[31:0]
                ((unsigned)(ga >> 32) & 0x01FFFFFFu) | (2u << 30)};  // [56:32] | type=2
    // group1: data_size=2B, tensor_dim0/1, tile_dim0/1, tensor_dim0_stride
    ull q0 = (1ull << 16) | (tdim0 << 48);
    ull q1 = (tdim0 >> 16) | (tdim1 << 16) | ((ull)tile_w << 48);
    ull q2 = (ull)tile_h | (row_stride_elems << 32);
    ull q3 = (row_stride_elems >> 32) & 0xFFFFull;
    union { ull q[4]; i32x8 v; } g1u;
    g1u.q[0] = q0; g1u.q[1] = q1; g1u.q[2] = q2; g1u.q[3] = q3;
    i32x4 z4 = {0, 0, 0, 0};
#if __clang_major__ >= 23
    i32x8 z8 = {0, 0, 0, 0, 0, 0, 0, 0};
    __builtin_amdgcn_tensor_load_to_lds(g0, g1u.v, z4, z4, z8, 0);
#else
    __builtin_amdgcn_tensor_load_to_lds(g0, g1u.v, z4, z4, 0);
#endif
}
#endif  // HAVE_TDM

// ---------------------------------------------------------------------------
// f32 -> f16 conversion (weights, input image)
// ---------------------------------------------------------------------------
__global__ void k_f32_to_f16(const float* __restrict__ src, _Float16* __restrict__ dst,
                             size_t n) {
    size_t i = (size_t)blockIdx.x * 256 + threadIdx.x;
    if (i < n) dst[i] = (_Float16)src[i];
}

// ---------------------------------------------------------------------------
// 3x3 SAME conv as implicit GEMM. One wave = 16 pixels x 16 Cout.
// K = 9*Cin swept in 288-element chunks (32 channels x 9 taps) so the
// (ci, tap) decomposition is compile-time after unrolling (no divisions).
// Weight tiles (16 x 288, row stride K) stream through LDS via TDM,
// double-buffered on TENSORcnt; activations are gathered into LDS once per
// chunk with precomputed window offsets, then both fragments are read with
// paired b32 LDS loads feeding v_wmma_f32_16x16x32_f16.
// ---------------------------------------------------------------------------
#define C3_KB 288   // 32 input channels * 9 taps

__global__ __launch_bounds__(32) void k_conv3x3(const _Float16* __restrict__ in,
                                                const _Float16* __restrict__ w,
                                                const float* __restrict__ bias,
                                                _Float16* __restrict__ out,
                                                int Cin, int Cout, int H, int W) {
    // smem layout: A tile [16][C3_KB] at 0; B ping/pong [16][C3_KB] each after.
    __shared__ _Float16 smem[3 * 16 * C3_KB];
    const int lane = threadIdx.x;
    const int hi = lane >> 4, l15 = lane & 15;
    const int pix0 = blockIdx.x * 16;
    const int co0  = blockIdx.y * 16;
    const int HW = H * W;
    const int pix = pix0 + l15;
    const int py = pix / W, px = pix - py * W;
    const bool mv = pix < HW;
    const int K = Cin * 9;
    const int n = co0 + l15;

    // per-lane 3x3 window validity + offsets (tap r = (dy+1)*3 + (dx+1))
    bool vm[9];
    int  ofs[9];
#pragma unroll
    for (int r = 0; r < 9; ++r) {
        const int dy = r / 3 - 1, dx = r - (r / 3) * 3 - 1;
        const int yy = py + dy, xx = px + dx;
        vm[r] = mv && (yy >= 0) && (yy < H) && (xx >= 0) && (xx < W);
        ofs[r] = yy * W + xx;
    }

    _Float16* sA = smem;                       // LDS byte offset 0
    const unsigned sB_off0   = (unsigned)(16 * C3_KB * 2);
    const unsigned sB_stride = (unsigned)(16 * C3_KB * 2);
    v8f acc = {};
    const int nkc = (K + C3_KB - 1) / C3_KB;
#if HAVE_TDM
    tdm_load_2d_f16(sB_off0, w + (size_t)co0 * K, C3_KB, 16, (ull)K, (ull)K, 16ull);
#endif
    for (int kc = 0; kc < nkc; ++kc) {
        const int buf = kc & 1;
        const int cbase = kc * 32;
        _Float16* sB = smem + 16 * C3_KB + buf * (16 * C3_KB);

        // --- stage A tile: lane (pixel=l15, half=hi) covers channels
        //     [cbase + hi*16, +16), taps unrolled (constant r) ---
        for (int ci2 = 0; ci2 < 16; ++ci2) {
            const int ci = cbase + hi * 16 + ci2;
            const bool cok = ci < Cin;     // uniform
            const _Float16* ip = in + (size_t)ci * HW;
#pragma unroll
            for (int r = 0; r < 9; ++r) {
                _Float16 val = (_Float16)0.f;
                if (cok && vm[r]) val = ip[ofs[r]];
                sA[l15 * C3_KB + (hi * 16 + ci2) * 9 + r] = val;
            }
        }
        if (kc + 1 < nkc)
            __builtin_prefetch(in + (size_t)(cbase + 32) * HW + ofs[4], 0, 1);

        // --- weight tile for this chunk (TDM double buffer) ---
#if HAVE_TDM
        if (kc + 1 < nkc) {
            const int k1 = (kc + 1) * C3_KB;
            tdm_load_2d_f16(sB_off0 + (buf ^ 1) * sB_stride, w + (size_t)co0 * K + k1,
                            C3_KB, 16, (ull)K, (ull)(K - k1), 16ull);
            WAIT_TC(1);   // newest still in flight; current buffer done
        } else {
            WAIT_TC(0);
        }
#else
        for (int idx = lane; idx < 16 * C3_KB; idx += 32) {
            const int rr = idx / C3_KB, cc = idx - rr * C3_KB;
            const int k = kc * C3_KB + cc;
            sB[idx] = (k < K) ? w[(size_t)(co0 + rr) * K + k] : (_Float16)0.f;
        }
#endif
        LDS_FENCE();
        // --- 9 WMMAs per chunk; both fragments use row-major [row][k] tiles ---
#pragma unroll
        for (int j = 0; j < 9; ++j) {
            v16h a, b;
#pragma unroll
            for (int v = 0; v < 8; ++v) {
                const int kb = j * 32 + ((v < 4) ? 0 : 16) + hi * 8 + (v & 3) * 2;
                ld2h(sA + l15 * C3_KB + kb, a, 2 * v);
                ld2h(sB + l15 * C3_KB + kb, b, 2 * v);
            }
            acc = __builtin_amdgcn_wmma_f32_16x16x32_f16(false, a, false, b, (short)0, acc,
                                                         false, false);
        }
        LDS_FENCE();   // keep next chunk's staging after this chunk's reads
    }
#pragma unroll
    for (int v = 0; v < 8; ++v) {
        const int mm = v + hi * 8;           // D row: VGPR v holds M=v / M=v+8
        const int opix = pix0 + mm;
        if (opix < HW) {
            float val = acc[v] + bias[n];
            out[(size_t)n * HW + opix] = (_Float16)(val > 0.f ? val : 0.f);
        }
    }
}

// ---------------------------------------------------------------------------
// 2x2 max-pool (VALID, floor), f16
// ---------------------------------------------------------------------------
__global__ void k_maxpool2(const _Float16* __restrict__ in, _Float16* __restrict__ out,
                           int C, int H, int W) {
    const int Ho = H >> 1, Wo = W >> 1;
    const int total = C * Ho * Wo;
    int i = blockIdx.x * 256 + threadIdx.x;
    if (i >= total) return;
    int c = i / (Ho * Wo), rp = i - c * (Ho * Wo);
    int y = rp / Wo, x = rp - y * Wo;
    const _Float16* p = in + (size_t)c * H * W + (2 * y) * W + 2 * x;
    float m0 = fmaxf((float)p[0], (float)p[1]);
    float m1 = fmaxf((float)p[W], (float)p[W + 1]);
    out[i] = (_Float16)fmaxf(m0, m1);
}

// ---------------------------------------------------------------------------
// 1x1 conv for RPN heads (tiny Cout: 18/36) - plain VALU
// ---------------------------------------------------------------------------
__global__ void k_conv1x1(const _Float16* __restrict__ in, const float* __restrict__ w,
                          const float* __restrict__ b, float* __restrict__ out, int Cout) {
    int i = blockIdx.x * 256 + threadIdx.x;
    if (i >= Cout * NPIX) return;
    int co = i / NPIX, p = i - co * NPIX;
    float s = b[co];
    for (int k = 0; k < 512; ++k) s += (float)in[(size_t)k * NPIX + p] * w[co * 512 + k];
    out[i] = s;
}

// ---------------------------------------------------------------------------
// anchors (normalized xyxy)
// ---------------------------------------------------------------------------
__global__ void k_anchors(float* __restrict__ anch) {
    int a = blockIdx.x * 256 + threadIdx.x;
    if (a >= NA) return;
    int pix = a / 9, t = a - pix * 9;
    int y = pix / FW, x = pix - y * FW;
    const float ratios[3] = {0.5f, 1.f, 2.f};
    const float scales[3] = {8.f, 16.f, 32.f};
    int r = t / 3, s = t - r * 3;
    float sq = sqrtf(ratios[r]);
    float hh = 16.f * scales[s] * sq;
    float ww = 16.f * scales[s] / sq;
    float cx = x * 16.f, cy = y * 16.f;
    anch[a * 4 + 0] = (cx - ww * 0.5f) / (float)IMG_W;
    anch[a * 4 + 1] = (cy - hh * 0.5f) / (float)IMG_H;
    anch[a * 4 + 2] = (cx + ww * 0.5f) / (float)IMG_W;
    anch[a * 4 + 3] = (cy + hh * 0.5f) / (float)IMG_H;
}

// ---------------------------------------------------------------------------
// RPN decode: softmax fg score, decode+clip boxes, validity, and also write
// the raw pred_rpn_cls / pred_rpn_reg outputs
// ---------------------------------------------------------------------------
__global__ void k_rpn_decode(const float* __restrict__ pc, const float* __restrict__ pr,
                             const float* __restrict__ anch, float* __restrict__ boxes,
                             float* __restrict__ scores, float* __restrict__ out) {
    int a = blockIdx.x * 256 + threadIdx.x;
    if (a >= NA) return;
    int pix = a / 9, t = a - pix * 9;
    float c0 = pc[(t * 2 + 0) * NPIX + pix];
    float c1 = pc[(t * 2 + 1) * NPIX + pix];
    out[O_RPN_CLS + a * 2 + 0] = c0;
    out[O_RPN_CLS + a * 2 + 1] = c1;
    float mx = fmaxf(c0, c1);
    float e0 = expf(c0 - mx), e1 = expf(c1 - mx);
    float s = e1 / (e0 + e1);
    float tr[4];
    for (int j = 0; j < 4; ++j) {
        tr[j] = pr[(t * 4 + j) * NPIX + pix];
        out[O_RPN_REG + a * 4 + j] = tr[j];
    }
    float ax1 = anch[a * 4 + 0], ay1 = anch[a * 4 + 1];
    float ax2 = anch[a * 4 + 2], ay2 = anch[a * 4 + 3];
    float aw = ax2 - ax1, ah = ay2 - ay1;
    float acx = (ax1 + ax2) * 0.5f, acy = (ay1 + ay2) * 0.5f;
    float cx = tr[0] * aw + acx, cy = tr[1] * ah + acy;
    float w = expf(tr[2]) * aw, h = expf(tr[3]) * ah;
    float x1 = fminf(fmaxf(cx - w * 0.5f, 0.f), 1.f);
    float y1 = fminf(fmaxf(cy - h * 0.5f, 0.f), 1.f);
    float x2 = fminf(fmaxf(cx + w * 0.5f, 0.f), 1.f);
    float y2 = fminf(fmaxf(cy + h * 0.5f, 0.f), 1.f);
    boxes[a * 4 + 0] = x1; boxes[a * 4 + 1] = y1;
    boxes[a * 4 + 2] = x2; boxes[a * 4 + 3] = y2;
    bool valid = ((y2 - y1) >= 0.016f) && ((x2 - x1) >= 0.016f);
    scores[a] = valid ? s : -2.f;   // invalid == pre-suppressed
}

// ---------------------------------------------------------------------------
// Greedy NMS: iterative global argmax == processing in descending score order
// ---------------------------------------------------------------------------
__global__ __launch_bounds__(256) void k_nms(const float* __restrict__ boxes,
                                             float* __restrict__ scores,
                                             float* __restrict__ rois) {
    __shared__ float sv[256];
    __shared__ int   si[256];
    __shared__ float kb[4];
    __shared__ int   kany;
    const int tid = threadIdx.x;
    for (int it = 0; it < NMS_K; ++it) {
        float bestv = -1e30f;
        int besti = -1;
        for (int i = tid; i < NA; i += 256) {
            float s = scores[i];
            if (s > bestv) { bestv = s; besti = i; }
        }
        sv[tid] = bestv; si[tid] = besti;
        __syncthreads();
        for (int off = 128; off > 0; off >>= 1) {
            if (tid < off && sv[tid + off] > sv[tid]) {
                sv[tid] = sv[tid + off]; si[tid] = si[tid + off];
            }
            __syncthreads();
        }
        if (tid == 0) {
            int idx = si[0];
            int any = (idx >= 0) && (sv[0] > -1.5f);
            kany = any;
            if (any) {
                kb[0] = boxes[idx * 4 + 0]; kb[1] = boxes[idx * 4 + 1];
                kb[2] = boxes[idx * 4 + 2]; kb[3] = boxes[idx * 4 + 3];
            } else {
                kb[0] = 0.f; kb[1] = 0.f; kb[2] = 0.1f; kb[3] = 0.1f;
            }
            rois[it * 4 + 0] = kb[0]; rois[it * 4 + 1] = kb[1];
            rois[it * 4 + 2] = kb[2]; rois[it * 4 + 3] = kb[3];
        }
        __syncthreads();
        if (kany) {
            const float bx1 = kb[0], by1 = kb[1], bx2 = kb[2], by2 = kb[3];
            const float barea = (bx2 - bx1) * (by2 - by1);
            for (int i = tid; i < NA; i += 256) {
                if (scores[i] > -1.5f) {
                    float x1 = boxes[i * 4 + 0], y1 = boxes[i * 4 + 1];
                    float x2 = boxes[i * 4 + 2], y2 = boxes[i * 4 + 3];
                    float ix1 = fmaxf(bx1, x1), iy1 = fmaxf(by1, y1);
                    float ix2 = fminf(bx2, x2), iy2 = fminf(by2, y2);
                    float iw = fmaxf(ix2 - ix1, 0.f), ih = fmaxf(iy2 - iy1, 0.f);
                    float inter = iw * ih;
                    float iou = inter / (barea + (x2 - x1) * (y2 - y1) - inter + 1e-12f);
                    if (iou > 0.7f) scores[i] = -2.f;   // includes self
                }
            }
        }
        __syncthreads();
    }
}

// ---------------------------------------------------------------------------
// RPN target maker (3 passes + sampling)
// ---------------------------------------------------------------------------
__global__ void k_rpn_t1(const float* __restrict__ anch, const float* __restrict__ gt,
                         float* __restrict__ ioub, float* __restrict__ ioumax,
                         int* __restrict__ iouarg, int* __restrict__ gmax) {
    int a = blockIdx.x * 256 + threadIdx.x;
    if (a >= NA) return;
    float x1 = anch[a * 4], y1 = anch[a * 4 + 1], x2 = anch[a * 4 + 2], y2 = anch[a * 4 + 3];
    bool inside = (x1 >= 0.f) && (y1 >= 0.f) && (x2 <= 1.f) && (y2 <= 1.f);
    float aarea = (x2 - x1) * (y2 - y1);
    float best = -1.f;
    int barg = 0;
    for (int g = 0; g < 8; ++g) {
        float gx1 = gt[g * 4], gy1 = gt[g * 4 + 1], gx2 = gt[g * 4 + 2], gy2 = gt[g * 4 + 3];
        float garea = (gx2 - gx1) * (gy2 - gy1);
        float ix1 = fmaxf(x1, gx1), iy1 = fmaxf(y1, gy1);
        float ix2 = fminf(x2, gx2), iy2 = fminf(y2, gy2);
        float iw = fmaxf(ix2 - ix1, 0.f), ih = fmaxf(iy2 - iy1, 0.f);
        float inter = iw * ih;
        float iou = inter / (aarea + garea - inter);
        iou = inside ? iou : -1.f;
        ioub[(size_t)a * 8 + g] = iou;
        if (iou > best) { best = iou; barg = g; }
        if (inside) atomicMax(&gmax[g], __float_as_int(iou));
    }
    ioumax[a] = best;
    iouarg[a] = barg;
}

__global__ void k_rpn_t2(const float* __restrict__ anch, const float* __restrict__ gt,
                         const float* __restrict__ ioub, const float* __restrict__ ioumax,
                         const int* __restrict__ iouarg, const int* __restrict__ gmax,
                         int* __restrict__ label, float* __restrict__ out) {
    int a = blockIdx.x * 256 + threadIdx.x;
    if (a >= NA) return;
    float x1 = anch[a * 4], y1 = anch[a * 4 + 1], x2 = anch[a * 4 + 2], y2 = anch[a * 4 + 3];
    bool inside = (x1 >= 0.f) && (y1 >= 0.f) && (x2 <= 1.f) && (y2 <= 1.f);
    int lab = -1;
    if (inside && ioumax[a] < 0.3f) lab = 0;
    bool best = false;
    for (int g = 0; g < 8; ++g)
        if (inside && __float_as_int(ioub[(size_t)a * 8 + g]) == gmax[g]) best = true;
    if (best) lab = 1;
    if (inside && ioumax[a] >= 0.7f) lab = 1;
    label[a] = lab;
    // regression target vs assigned gt
    int g = iouarg[a];
    float gx1 = gt[g * 4], gy1 = gt[g * 4 + 1], gx2 = gt[g * 4 + 2], gy2 = gt[g * 4 + 3];
    float gw = gx2 - gx1, gh = gy2 - gy1;
    float gcx = (gx1 + gx2) * 0.5f, gcy = (gy1 + gy2) * 0.5f;
    float aw = x2 - x1, ah = y2 - y1;
    float acx = (x1 + x2) * 0.5f, acy = (y1 + y2) * 0.5f;
    float t0 = (gcx - acx) / aw, t1 = (gcy - acy) / ah;
    float t2 = logf(gw / aw), t3 = logf(gh / ah);
    if (!inside) { t0 = t1 = t2 = t3 = 0.f; }
    out[O_TG_RPN_REG + a * 4 + 0] = t0;
    out[O_TG_RPN_REG + a * 4 + 1] = t1;
    out[O_TG_RPN_REG + a * 4 + 2] = t2;
    out[O_TG_RPN_REG + a * 4 + 3] = t3;
}

__global__ void k_rpn_t3a(int* __restrict__ label, int* __restrict__ cnt) {
    int a = blockIdx.x * 256 + threadIdx.x;
    if (a >= NA) return;
    if (label[a] == 1) {
        int r = atomicAdd(&cnt[0], 1);
        if (r >= 128) label[a] = -1;
    }
}

__global__ void k_rpn_t3b(int* __restrict__ label, int* __restrict__ cnt,
                          float* __restrict__ out) {
    int a = blockIdx.x * 256 + threadIdx.x;
    if (a >= NA) return;
    int np = cnt[0];
    if (np > 128) np = 128;
    int cap = 256 - np;
    int lab = label[a];
    if (lab == 0) {
        int r = atomicAdd(&cnt[1], 1);
        if (r >= cap) { lab = -1; label[a] = -1; }
    }
    out[O_TG_RPN_CLS + a] = (float)lab;
}

// ---------------------------------------------------------------------------
// Fast R-CNN target maker (deterministic first-k sampling)
// ---------------------------------------------------------------------------
__global__ __launch_bounds__(256) void k_frcnn_tgt(const float* __restrict__ rois,
                                                   const float* __restrict__ gt,
                                                   const int* __restrict__ glabel,
                                                   float* __restrict__ riomax,
                                                   int* __restrict__ rioarg,
                                                   float* __restrict__ srois,
                                                   int* __restrict__ tgcls_i,
                                                   float* __restrict__ out) {
    const int tid = threadIdx.x;
    for (int i = tid; i < RDIM; i += 256) {
        float x1, y1, x2, y2;
        if (i < NMS_K) {
            x1 = rois[i * 4]; y1 = rois[i * 4 + 1]; x2 = rois[i * 4 + 2]; y2 = rois[i * 4 + 3];
        } else {
            int g = i - NMS_K;
            x1 = gt[g * 4]; y1 = gt[g * 4 + 1]; x2 = gt[g * 4 + 2]; y2 = gt[g * 4 + 3];
        }
        float aarea = (x2 - x1) * (y2 - y1);
        float best = -1.f;
        int barg = 0;
        for (int g = 0; g < 8; ++g) {
            float gx1 = gt[g * 4], gy1 = gt[g * 4 + 1], gx2 = gt[g * 4 + 2], gy2 = gt[g * 4 + 3];
            float garea = (gx2 - gx1) * (gy2 - gy1);
            float ix1 = fmaxf(x1, gx1), iy1 = fmaxf(y1, gy1);
            float ix2 = fminf(x2, gx2), iy2 = fminf(y2, gy2);
            float iw = fmaxf(ix2 - ix1, 0.f), ih = fmaxf(iy2 - iy1, 0.f);
            float inter = iw * ih;
            float iou = inter / (aarea + garea - inter + 1e-12f);
            if (iou > best) { best = iou; barg = g; }
        }
        riomax[i] = best;
        rioarg[i] = barg;
    }
    __syncthreads();
    if (tid == 0) {
        int npos = 0;
        for (int i = 0; i < RDIM; ++i)
            if (riomax[i] >= 0.5f) npos++;
        int n_pos = npos < 32 ? npos : 32;
        int keep[NSAMPLE];
        int kp = 0;
        for (int i = 0; i < RDIM && kp < n_pos; ++i)
            if (riomax[i] >= 0.5f) keep[kp++] = i;
        int kn = 0;
        for (int i = 0; i < RDIM && kn < NSAMPLE - n_pos; ++i)
            if (riomax[i] < 0.5f) keep[n_pos + kn++] = i;
        for (int s = 0; s < NSAMPLE; ++s) {
            int i = keep[s];
            float x1, y1, x2, y2;
            if (i < NMS_K) {
                x1 = rois[i * 4]; y1 = rois[i * 4 + 1]; x2 = rois[i * 4 + 2]; y2 = rois[i * 4 + 3];
            } else {
                int g = i - NMS_K;
                x1 = gt[g * 4]; y1 = gt[g * 4 + 1]; x2 = gt[g * 4 + 2]; y2 = gt[g * 4 + 3];
            }
            int cls = (s < n_pos) ? (glabel[rioarg[i]] + 1) : 0;
            tgcls_i[s] = cls;
            out[O_TG_CLS + s] = (float)cls;
            srois[s * 4 + 0] = x1; srois[s * 4 + 1] = y1;
            srois[s * 4 + 2] = x2; srois[s * 4 + 3] = y2;
            int g = rioarg[i];
            float gx1 = gt[g * 4], gy1 = gt[g * 4 + 1], gx2 = gt[g * 4 + 2], gy2 = gt[g * 4 + 3];
            float gw = gx2 - gx1, gh = gy2 - gy1;
            float gcx = (gx1 + gx2) * 0.5f, gcy = (gy1 + gy2) * 0.5f;
            float rw = x2 - x1, rh = y2 - y1;
            float rcx = (x1 + x2) * 0.5f, rcy = (y1 + y2) * 0.5f;
            out[O_TG_REG + s * 4 + 0] = ((gcx - rcx) / rw) / 0.1f;
            out[O_TG_REG + s * 4 + 1] = ((gcy - rcy) / rh) / 0.1f;
            out[O_TG_REG + s * 4 + 2] = logf(gw / rw) / 0.2f;
            out[O_TG_REG + s * 4 + 3] = logf(gh / rh) / 0.2f;
        }
    }
}

// ---------------------------------------------------------------------------
// Caffe-style RoI max-pool to 7x7 (f16 feats), flattened for FC1
// ---------------------------------------------------------------------------
__global__ __launch_bounds__(128) void k_roi_pool(const _Float16* __restrict__ feat,
                                                  const float* __restrict__ srois,
                                                  _Float16* __restrict__ z0) {
    const int r = blockIdx.x;
    float rx1 = srois[r * 4 + 0] * FW, ry1 = srois[r * 4 + 1] * FH;
    float rx2 = srois[r * 4 + 2] * FW, ry2 = srois[r * 4 + 3] * FH;
    int x1 = (int)rintf(rx1), y1 = (int)rintf(ry1);
    int x2 = (int)rintf(rx2), y2 = (int)rintf(ry2);
    float rw = fmaxf((float)(x2 - x1 + 1), 1.f);
    float rh = fmaxf((float)(y2 - y1 + 1), 1.f);
    float bw = rw / 7.f, bh = rh / 7.f;
    for (int c = threadIdx.x; c < 512; c += 128) {
        const _Float16* f = feat + (size_t)c * NPIX;
        for (int py = 0; py < 7; ++py) {
            int hs = iclampi((int)floorf(py * bh) + y1, 0, FH);
            int he = iclampi((int)ceilf((py + 1) * bh) + y1, 0, FH);
            for (int px = 0; px < 7; ++px) {
                int ws = iclampi((int)floorf(px * bw) + x1, 0, FW);
                int we = iclampi((int)ceilf((px + 1) * bw) + x1, 0, FW);
                float m = -1e30f;
                bool any = false;
                for (int yy = hs; yy < he; ++yy)
                    for (int xx = ws; xx < we; ++xx) {
                        m = fmaxf(m, (float)f[yy * FW + xx]);
                        any = true;
                    }
                z0[(size_t)r * 25088 + c * 49 + py * 7 + px] = (_Float16)(any ? m : 0.f);
            }
        }
    }
}

// ---------------------------------------------------------------------------
// FC GEMM: C(MxN) = relu(A(MxK) * B(KxN) + bias). One wave per 16x16 tile.
// K staged through LDS in 256-element chunks, double-buffered with TDM,
// pipelined via TENSORcnt.
// ---------------------------------------------------------------------------
#define FC_KB 256

__global__ __launch_bounds__(32) void k_fc_wmma(const _Float16* __restrict__ A,
                                                const _Float16* __restrict__ B,
                                                const float* __restrict__ bias,
                                                _Float16* __restrict__ C,
                                                int M, int K, int N, int relu) {
    // LDS: A ping/pong (16xFC_KB each) then B ping/pong (FC_KBx16 each).
    __shared__ _Float16 lds[2 * 16 * FC_KB + 2 * FC_KB * 16];
    const int lane = threadIdx.x;
    const int hi = lane >> 4, l15 = lane & 15;
    const int m0 = blockIdx.y * 16, n0 = blockIdx.x * 16;
    v8f acc = {};
    const int nk = K / FC_KB;   // exact for our shapes (25088, 4096)
#if HAVE_TDM
    const unsigned offA0 = 0u;
    const unsigned offStrideA = (unsigned)(16 * FC_KB * 2);
    const unsigned offB0 = (unsigned)(2 * 16 * FC_KB * 2);
    const unsigned offStrideB = (unsigned)(FC_KB * 16 * 2);
    tdm_load_2d_f16(offA0, A + (size_t)m0 * K, FC_KB, 16, (ull)K, (ull)K, 16ull);
    tdm_load_2d_f16(offB0, B + n0, 16, FC_KB, (ull)N, (ull)N, (ull)FC_KB);
#endif
    for (int kc = 0; kc < nk; ++kc) {
        const int buf = kc & 1;
        _Float16* curA = lds + buf * (16 * FC_KB);
        _Float16* curB = lds + 2 * 16 * FC_KB + buf * (FC_KB * 16);
#if HAVE_TDM
        if (kc + 1 < nk) {
            const int nb = buf ^ 1;
            const int k1 = (kc + 1) * FC_KB;
            tdm_load_2d_f16(offA0 + nb * offStrideA, A + (size_t)m0 * K + k1, FC_KB, 16,
                            (ull)K, (ull)K, 16ull);
            tdm_load_2d_f16(offB0 + nb * offStrideB, B + (size_t)k1 * N + n0, 16, FC_KB,
                            (ull)N, (ull)N, (ull)FC_KB);
            WAIT_TC(2);   // two newest still in flight; current buffer's pair done
        } else {
            WAIT_TC(0);
        }
#else
        for (int i = lane; i < 16 * FC_KB; i += 32) {
            int rr = i / FC_KB, cc = i - rr * FC_KB;
            curA[i] = A[(size_t)(m0 + rr) * K + kc * FC_KB + cc];
        }
        for (int i = lane; i < FC_KB * 16; i += 32) {
            int rr = i / 16, cc = i - rr * 16;
            curB[i] = B[(size_t)(kc * FC_KB + rr) * N + n0 + cc];
        }
#endif
        LDS_FENCE();
#pragma unroll
        for (int kk = 0; kk < FC_KB / 32; ++kk) {
            v16h a, b;
#pragma unroll
            for (int v = 0; v < 8; ++v) {
                const int kb = kk * 32 + ((v < 4) ? 0 : 16) + hi * 8 + (v & 3) * 2;
                ld2h(curA + l15 * FC_KB + kb, a, 2 * v);   // contiguous pair
                b[2 * v + 0] = curB[(kb + 0) * 16 + l15];  // stride-16: scalar
                b[2 * v + 1] = curB[(kb + 1) * 16 + l15];
            }
            acc = __builtin_amdgcn_wmma_f32_16x16x32_f16(false, a, false, b, (short)0, acc,
                                                         false, false);
        }
        LDS_FENCE();   // keep next chunk's TDM writes after this chunk's reads
    }
#pragma unroll
    for (int v = 0; v < 8; ++v) {
        const int mm = v + hi * 8;
        float val = acc[v] + bias[n0 + l15];
        if (relu) val = val > 0.f ? val : 0.f;
        C[(size_t)(m0 + mm) * N + n0 + l15] = (_Float16)val;
    }
}

// ---------------------------------------------------------------------------
// Final cls/reg heads (4096 -> 21, 4096 -> 84) with target-class gather
// ---------------------------------------------------------------------------
__global__ __launch_bounds__(128) void k_heads(const _Float16* __restrict__ z,
                                               const float* __restrict__ Wcl,
                                               const float* __restrict__ bcl,
                                               const float* __restrict__ Wrg,
                                               const float* __restrict__ brg,
                                               const int* __restrict__ tgcls,
                                               float* __restrict__ out) {
    __shared__ float regall[84];
    const int r = blockIdx.x, tid = threadIdx.x;
    const _Float16* zr = z + (size_t)r * 4096;
    for (int o = tid; o < 105; o += 128) {
        if (o < 21) {
            float s = bcl[o];
            for (int k = 0; k < 4096; ++k) s += (float)zr[k] * Wcl[(size_t)k * 21 + o];
            out[O_PRED_CLS + r * 21 + o] = s;
        } else {
            int j = o - 21;
            float s = brg[j];
            for (int k = 0; k < 4096; ++k) s += (float)zr[k] * Wrg[(size_t)k * 84 + j];
            regall[j] = s;
        }
    }
    __syncthreads();
    if (tid < 4) {
        int tc = tgcls[r];
        out[O_PRED_REG + r * 4 + tid] = regall[tc * 4 + tid];
    }
}

// ---------------------------------------------------------------------------
// host orchestration
// ---------------------------------------------------------------------------
extern "C" void kernel_launch(void* const* d_in, const int* in_sizes, int n_in,
                              void* d_out, int out_size, void* d_ws, size_t ws_size,
                              hipStream_t stream) {
    (void)in_sizes; (void)n_in; (void)out_size; (void)ws_size;
    const float* x      = (const float*)d_in[0];
    const float* bbox   = (const float*)d_in[1];
    const int*   glabel = (const int*)d_in[2];
    const float* vggW[13];
    const float* vggB[13];
    for (int i = 0; i < 13; ++i) {
        vggW[i] = (const float*)d_in[3 + 2 * i];
        vggB[i] = (const float*)d_in[4 + 2 * i];
    }
    const float* rpnWi = (const float*)d_in[29];
    const float* rpnBi = (const float*)d_in[30];
    const float* rpnWc = (const float*)d_in[31];
    const float* rpnBc = (const float*)d_in[32];
    const float* rpnWr = (const float*)d_in[33];
    const float* rpnBr = (const float*)d_in[34];
    const float* W1  = (const float*)d_in[35];
    const float* b1  = (const float*)d_in[36];
    const float* W2  = (const float*)d_in[37];
    const float* b2  = (const float*)d_in[38];
    const float* Wcl = (const float*)d_in[39];
    const float* bcl = (const float*)d_in[40];
    const float* Wrg = (const float*)d_in[41];
    const float* brg = (const float*)d_in[42];
    float* out = (float*)d_out;

    // bump allocator over workspace
    char* p = (char*)d_ws;
    auto alloc = [&](size_t bytes) -> void* {
        char* r = p;
        p += (bytes + 255) & ~(size_t)255;
        return (void*)r;
    };
    auto cvt = [&](const float* src, _Float16* dst, size_t n) {
        k_f32_to_f16<<<(unsigned)((n + 255) / 256), 256, 0, stream>>>(src, dst, n);
    };

    _Float16* actA = (_Float16*)alloc((size_t)64 * IMG_H * IMG_W * 2);
    _Float16* actB = (_Float16*)alloc((size_t)64 * IMG_H * IMG_W * 2);

    const int convCout[13] = {64, 64, 128, 128, 256, 256, 256, 512, 512, 512, 512, 512, 512};
    const int convCin[13]  = {3, 64, 64, 128, 128, 256, 256, 256, 512, 512, 512, 512, 512};
    _Float16* wf[13];
    for (int i = 0; i < 13; ++i) {
        size_t ne = (size_t)convCout[i] * convCin[i] * 9;
        wf[i] = (_Float16*)alloc(ne * 2);
        cvt(vggW[i], wf[i], ne);
    }
    _Float16* wrpn = (_Float16*)alloc((size_t)512 * 512 * 9 * 2);
    cvt(rpnWi, wrpn, (size_t)512 * 512 * 9);
    _Float16* wfc1 = (_Float16*)alloc((size_t)25088 * 4096 * 2);
    cvt(W1, wfc1, (size_t)25088 * 4096);
    _Float16* wfc2 = (_Float16*)alloc((size_t)4096 * 4096 * 2);
    cvt(W2, wfc2, (size_t)4096 * 4096);

    float* pc     = (float*)alloc((size_t)18 * NPIX * 4);
    float* pr     = (float*)alloc((size_t)36 * NPIX * 4);
    float* anch   = (float*)alloc((size_t)NA * 4 * 4);
    float* boxes  = (float*)alloc((size_t)NA * 4 * 4);
    float* scores = (float*)alloc((size_t)NA * 4);
    float* ioub   = (float*)alloc((size_t)NA * 8 * 4);
    float* ioumax = (float*)alloc((size_t)NA * 4);
    int*   iouarg = (int*)alloc((size_t)NA * 4);
    int*   labels = (int*)alloc((size_t)NA * 4);
    int*   gmax   = (int*)alloc(8 * 4);
    int*   cnt    = (int*)alloc(2 * 4);
    float* rois   = (float*)alloc((size_t)NMS_K * 4 * 4);
    float* riomax = (float*)alloc((size_t)RDIM * 4);
    int*   rioarg = (int*)alloc((size_t)RDIM * 4);
    float* srois  = (float*)alloc((size_t)NSAMPLE * 4 * 4);
    int*   tgclsi = (int*)alloc((size_t)NSAMPLE * 4);
    _Float16* z0  = (_Float16*)alloc((size_t)NSAMPLE * 25088 * 2);
    _Float16* z1  = (_Float16*)alloc((size_t)NSAMPLE * 4096 * 2);
    _Float16* z2  = (_Float16*)alloc((size_t)NSAMPLE * 4096 * 2);

    // --- VGG tower ---
    cvt(x, actA, (size_t)3 * IMG_H * IMG_W);
    _Float16* cur = actA;
    _Float16* oth = actB;
    int H = IMG_H, W = IMG_W, cin = 3, li = 0;
    const int cfg[17] = {64, 64, -1, 128, 128, -1, 256, 256, 256, -1,
                         512, 512, 512, -1, 512, 512, 512};
    for (int s = 0; s < 17; ++s) {
        if (cfg[s] < 0) {
            int Ho = H / 2, Wo = W / 2;
            int total = cin * Ho * Wo;
            k_maxpool2<<<(total + 255) / 256, 256, 0, stream>>>(cur, oth, cin, H, W);
            H = Ho; W = Wo;
            _Float16* t = cur; cur = oth; oth = t;
        } else {
            int cout = cfg[s];
            dim3 g((H * W + 15) / 16, cout / 16);
            k_conv3x3<<<g, 32, 0, stream>>>(cur, wf[li], vggB[li], oth, cin, cout, H, W);
            cin = cout; li++;
            _Float16* t = cur; cur = oth; oth = t;
        }
    }
    // cur = feats f16 (512, 37, 50)

    // --- RPN ---
    {
        dim3 g((NPIX + 15) / 16, 512 / 16);
        k_conv3x3<<<g, 32, 0, stream>>>(cur, wrpn, rpnBi, oth, 512, 512, FH, FW);
    }
    k_conv1x1<<<(18 * NPIX + 255) / 256, 256, 0, stream>>>(oth, rpnWc, rpnBc, pc, 18);
    k_conv1x1<<<(36 * NPIX + 255) / 256, 256, 0, stream>>>(oth, rpnWr, rpnBr, pr, 36);

    const int gNA = (NA + 255) / 256;
    k_anchors<<<gNA, 256, 0, stream>>>(anch);
    k_rpn_decode<<<gNA, 256, 0, stream>>>(pc, pr, anch, boxes, scores, out);
    k_nms<<<1, 256, 0, stream>>>(boxes, scores, rois);

    // --- RPN targets ---
    (void)hipMemsetAsync(gmax, 0, 8 * sizeof(int), stream);
    (void)hipMemsetAsync(cnt, 0, 2 * sizeof(int), stream);
    k_rpn_t1<<<gNA, 256, 0, stream>>>(anch, bbox, ioub, ioumax, iouarg, gmax);
    k_rpn_t2<<<gNA, 256, 0, stream>>>(anch, bbox, ioub, ioumax, iouarg, gmax, labels, out);
    k_rpn_t3a<<<gNA, 256, 0, stream>>>(labels, cnt);
    k_rpn_t3b<<<gNA, 256, 0, stream>>>(labels, cnt, out);

    // --- Fast R-CNN targets + RoI pool + FC head ---
    k_frcnn_tgt<<<1, 256, 0, stream>>>(rois, bbox, glabel, riomax, rioarg, srois, tgclsi, out);
    k_roi_pool<<<NSAMPLE, 128, 0, stream>>>(cur, srois, z0);
    {
        dim3 g1(4096 / 16, NSAMPLE / 16);
        k_fc_wmma<<<g1, 32, 0, stream>>>(z0, wfc1, b1, z1, NSAMPLE, 25088, 4096, 1);
        k_fc_wmma<<<g1, 32, 0, stream>>>(z1, wfc2, b2, z2, NSAMPLE, 4096, 4096, 1);
    }
    k_heads<<<NSAMPLE, 128, 0, stream>>>(z2, Wcl, bcl, Wrg, brg, tgclsi, out);
}
